// GPT2AttentionFlex_81458349735957
// MI455X (gfx1250) — compile-verified, matrix operand order checked by reading, not run on hardware
//
#include <hip/hip_runtime.h>

// ---------------------------------------------------------------------------
// GPT-2 attention block on gfx1250 (MI455X), wave32 + WMMA bf16.
//   B=2, S=2048, D=2048, H=16, HD=128
// Pipeline:
//   [cvt f32->bf16] -> [QKV GEMM wmma, head-split epilogue, V transposed]
//   -> [flash attention per (b,h,qtile) wave, wmma QK^T and PV]
//   -> [proj GEMM wmma, f32 out]
// GEMM: block tile 32x256, 8 waves (2x4), wave tile 16x64, LDS staged,
//       A tile filled via global_load_async_to_lds_b128 when available.
// Attention: batched operand loads (8 K-ops / 8 V-ops) for deep pipelining.
// ---------------------------------------------------------------------------

typedef __attribute__((ext_vector_type(16))) __bf16 bf16x16;
typedef __attribute__((ext_vector_type(8)))  float  floatx8;
typedef __attribute__((ext_vector_type(4)))  int    int4v;

#if __has_builtin(__builtin_amdgcn_global_load_async_to_lds_b128)
#define HAVE_ASYNC_LDS 1
#else
#define HAVE_ASYNC_LDS 0
#endif

static __device__ inline void wait_asynccnt0() {
#if __has_builtin(__builtin_amdgcn_s_wait_asynccnt)
    __builtin_amdgcn_s_wait_asynccnt(0);
#else
    asm volatile("s_wait_asynccnt 0" ::: "memory");
#endif
}

static __device__ inline __bf16 us2bf(unsigned short u) {
    union { unsigned short u; __bf16 b; } t; t.u = u; return t.b;
}

static __device__ inline unsigned short f2us_bf16(float f) {
    unsigned int x = __float_as_uint(f);
    unsigned int r = 0x7FFFu + ((x >> 16) & 1u);   // round-to-nearest-even
    x += r;
    return (unsigned short)(x >> 16);
}

static __device__ inline floatx8 wmma_bf16(bf16x16 a, bf16x16 b, floatx8 c) {
    return __builtin_amdgcn_wmma_f32_16x16x32_bf16(
        false, a, false, b, (short)0, c, false, false);
}

// ---------------------------------------------------------------------------
// f32 -> bf16 conversion (grid-strided)
// ---------------------------------------------------------------------------
__global__ void cvt_f32_bf16(const float* __restrict__ src,
                             unsigned short* __restrict__ dst, int n) {
    int i = blockIdx.x * blockDim.x + threadIdx.x;
    int stride = gridDim.x * blockDim.x;
    for (; i < n; i += stride) dst[i] = f2us_bf16(src[i]);
}

// ---------------------------------------------------------------------------
// bf16 WMMA GEMM:  C[M,N] = A[M,K] * W[K,N] + bias
//   block = 256 threads = 8 waves (2Mx4N); wave tile 16x64 (4 accumulators)
//   LDS: As[32][32] row-major (async-filled), Bs[256][32] transposed [n][k]
// MODE 0: scatter bf16 Q,K -> [B,H,S,HD]; V -> [B,H,HD,S] (transposed)
// MODE 1: write f32 [M,N]
// ---------------------------------------------------------------------------
template <int MODE>
__global__ __launch_bounds__(256) void gemm_bf16(
    const unsigned short* __restrict__ A,
    const unsigned short* __restrict__ W,
    const float* __restrict__ bias,
    int M, int K, int N,
    unsigned short* __restrict__ Qo,
    unsigned short* __restrict__ Ko,
    unsigned short* __restrict__ Vo,
    float* __restrict__ Fo) {
    __shared__ unsigned short As[32 * 32];    //  2 KB
    __shared__ unsigned short Bs[256 * 32];   // 16 KB, [n][k]

    const int tid    = threadIdx.x;
    const int lane   = tid & 31;
    const int wave   = tid >> 5;
    const int wave_m = wave >> 2;             // 0..1
    const int wave_n = wave & 3;              // 0..3
    const int ma     = blockIdx.y * 32;
    const int nb     = blockIdx.x * 256;
    const int row    = lane & 15;
    const int half   = lane >> 4;

    const int a_row = tid >> 3;               // manual path: 0..31
    const int a_c0  = (tid & 7) * 4;
    const int b_kk  = tid >> 3;               // 0..31
    const int b_n0  = (tid & 7) * 32;         // 0..224

    floatx8 acc[4];
    const floatx8 fz = {0.f, 0.f, 0.f, 0.f, 0.f, 0.f, 0.f, 0.f};
#pragma unroll
    for (int t = 0; t < 4; ++t) acc[t] = fz;

    for (int kc = 0; kc < K; kc += 32) {
        // ---- stage A tile 32x32 bf16 (2 KB)
#if HAVE_ASYNC_LDS
        if (tid < 128) {                      // 128 lanes x 16B = whole tile
            const int r2 = tid >> 2;          // 0..31
            const int c8 = (tid & 3) * 8;     // 0..24
            int4v* gp = (int4v*)(A + (size_t)(ma + r2) * K + kc + c8);
            int4v* lp = (int4v*)&As[r2 * 32 + c8];
            __builtin_amdgcn_global_load_async_to_lds_b128(gp, lp, 0, 0);
        }
#else
        {
            const unsigned short* ap = A + (size_t)(ma + a_row) * K + kc + a_c0;
#pragma unroll
            for (int j = 0; j < 4; ++j) As[a_row * 32 + a_c0 + j] = ap[j];
        }
#endif
        // ---- stage B tile 32x256 transposed into Bs[n][k]
        {
            const unsigned short* wp = W + (size_t)(kc + b_kk) * N + nb + b_n0;
#pragma unroll
            for (int j = 0; j < 32; ++j) Bs[(b_n0 + j) * 32 + b_kk] = wp[j];
        }
        if (kc + 32 < K) {  // prefetch next tiles (global_prefetch_b8)
            __builtin_prefetch(A + (size_t)(ma + a_row) * K + kc + 32 + a_c0, 0, 1);
            __builtin_prefetch(W + (size_t)(kc + 32 + b_kk) * N + nb + b_n0, 0, 1);
        }
#if HAVE_ASYNC_LDS
        wait_asynccnt0();
#endif
        __syncthreads();

        // ---- A operand (16x32), reused across 4 WMMAs
        bf16x16 a;
        const int abase = (wave_m * 16 + row) * 32 + half * 8;
#pragma unroll
        for (int i = 0; i < 8; ++i) {
            a[i]     = us2bf(As[abase + i]);
            a[8 + i] = us2bf(As[abase + 16 + i]);
        }
        bf16x16 bop[4];
#pragma unroll
        for (int t = 0; t < 4; ++t) {
            const int bbase = (wave_n * 64 + t * 16 + row) * 32 + half * 8;
#pragma unroll
            for (int i = 0; i < 8; ++i) {
                bop[t][i]     = us2bf(Bs[bbase + i]);
                bop[t][8 + i] = us2bf(Bs[bbase + 16 + i]);
            }
        }
#pragma unroll
        for (int t = 0; t < 4; ++t) acc[t] = wmma_bf16(a, bop[t], acc[t]);
        __syncthreads();
    }

    // ---- epilogue
    const int m_base = ma + wave_m * 16;
#pragma unroll
    for (int t = 0; t < 4; ++t) {
        const int n  = nb + wave_n * 64 + t * 16 + row;
        const float bv = bias[n];
#pragma unroll
        for (int r = 0; r < 8; ++r) {
            const int m = m_base + r + half * 8;
            const float v = acc[t][r] + bv;
            if (MODE == 0) {
                const int which = n >> 11;      // 0:q 1:k 2:v
                const int nm = n & 2047;
                const int h  = nm >> 7;
                const int d  = nm & 127;
                const int bi = m >> 11;
                const int s  = m & 2047;
                size_t idx;
                if (which == 2)  // V transposed: [B,H,HD,S]
                    idx = (((size_t)bi * 16 + h) * 128 + d) * 2048 + s;
                else             // Q,K: [B,H,S,HD]
                    idx = (((size_t)bi * 16 + h) * 2048 + s) * 128 + d;
                unsigned short* dst = (which == 0) ? Qo : (which == 1) ? Ko : Vo;
                dst[idx] = f2us_bf16(v);
            } else {
                Fo[(size_t)m * N + n] = v;
            }
        }
    }
}

// ---------------------------------------------------------------------------
// Flash attention: one wave per (b,h, 16-query tile). HD=128, S=2048.
//   Q,K: [B,H,S,HD]  V: [B,H,HD,S] (transposed -> contiguous PV operand)
//   Operand loads batched (8 at a time) so WMMAs issue back-to-back.
// ---------------------------------------------------------------------------
__global__ __launch_bounds__(256) void attn_flash(
    const unsigned short* __restrict__ Qb,
    const unsigned short* __restrict__ Kb,
    const unsigned short* __restrict__ Vt,
    unsigned short* __restrict__ Ctx) {
    __shared__ unsigned short Ps[8][16 * 32];   // per-wave P staging

    const int lane = threadIdx.x & 31;
    const int wave = threadIdx.x >> 5;
    const int gw   = blockIdx.x * 8 + wave;     // 0..4095
    const int bh   = gw >> 7;                   // 0..31  (b*16+h)
    const int qt   = gw & 127;
    const int q0   = qt * 16;
    const int row  = lane & 15;
    const int half = lane >> 4;

    const size_t base = (size_t)bh * 2048 * 128;
    const unsigned short* Qp = Qb + base;
    const unsigned short* Kp = Kb + base;
    const unsigned short* Vp = Vt + base;       // [HD][S]

    // Q tile in registers: 4 A-operands covering HD=128
    bf16x16 qa[4];
#pragma unroll
    for (int c = 0; c < 4; ++c) {
        const unsigned short* qp =
            Qp + (size_t)(q0 + row) * 128 + c * 32 + half * 8;
#pragma unroll
        for (int i = 0; i < 8; ++i) {
            qa[c][i]     = us2bf(qp[i]);
            qa[c][8 + i] = us2bf(qp[16 + i]);
        }
    }

    const floatx8 fz = {0.f, 0.f, 0.f, 0.f, 0.f, 0.f, 0.f, 0.f};
    floatx8 ctx[8];
#pragma unroll
    for (int t = 0; t < 8; ++t) ctx[t] = fz;
    float mrow[8], lrow[8];
#pragma unroll
    for (int r = 0; r < 8; ++r) { mrow[r] = -3.0e38f; lrow[r] = 0.f; }

    const float scale = 0.08838834764831845f;   // 1/sqrt(128)
    const int kend = q0 + 16;

    for (int kb = 0; kb < kend; kb += 32) {
        if (kb + 32 < kend) {  // prefetch next key/value chunks
            __builtin_prefetch(Kp + (size_t)(kb + 32 + lane) * 128, 0, 1);
            __builtin_prefetch(Vp + (size_t)lane * 2048 + kb + 32, 0, 1);
        }

        // ---- batch-load all 8 K operands (2 key tiles x 4 HD chunks)
        bf16x16 bk[8];
#pragma unroll
        for (int t = 0; t < 2; ++t) {
            const int key = kb + t * 16 + row;
#pragma unroll
            for (int c = 0; c < 4; ++c) {
                const unsigned short* kp =
                    Kp + (size_t)key * 128 + c * 32 + half * 8;
#pragma unroll
                for (int i = 0; i < 8; ++i) {
                    bk[t * 4 + c][i]     = us2bf(kp[i]);
                    bk[t * 4 + c][8 + i] = us2bf(kp[16 + i]);
                }
            }
        }
        // ---- scores: 8 WMMAs back-to-back
        floatx8 sc[2];
        sc[0] = fz; sc[1] = fz;
#pragma unroll
        for (int t = 0; t < 2; ++t)
#pragma unroll
            for (int c = 0; c < 4; ++c)
                sc[t] = wmma_bf16(qa[c], bk[t * 4 + c], sc[t]);

        // ---- causal mask + online softmax, P staged into LDS (A layout)
#pragma unroll
        for (int r = 0; r < 8; ++r) {
            const int qrow = q0 + r + half * 8;
            float s0 = sc[0][r] * scale;
            float s1 = sc[1][r] * scale;
            if (kb + row > qrow)      s0 = -3.0e38f;
            if (kb + 16 + row > qrow) s1 = -3.0e38f;

            float cmax = fmaxf(s0, s1);
#pragma unroll
            for (int off = 1; off < 16; off <<= 1)
                cmax = fmaxf(cmax, __shfl_xor(cmax, off, 32));
            const float mnew = fmaxf(mrow[r], cmax);

            const float p0 = __expf(s0 - mnew);
            const float p1 = __expf(s1 - mnew);
            float psum = p0 + p1;
#pragma unroll
            for (int off = 1; off < 16; off <<= 1)
                psum += __shfl_xor(psum, off, 32);

            const float alpha = __expf(mrow[r] - mnew);
            lrow[r] = lrow[r] * alpha + psum;
            mrow[r] = mnew;
#pragma unroll
            for (int t = 0; t < 8; ++t) ctx[t][r] *= alpha;

            Ps[wave][(r + half * 8) * 32 + row]      = f2us_bf16(p0);
            Ps[wave][(r + half * 8) * 32 + 16 + row] = f2us_bf16(p1);
        }
        asm volatile("s_wait_dscnt 0" ::: "memory");
        __builtin_amdgcn_wave_barrier();

        // ---- read P back as 16x32 bf16 A operand
        bf16x16 pa;
        const int pbase = row * 32 + half * 8;
#pragma unroll
        for (int i = 0; i < 8; ++i) {
            pa[i]     = us2bf(Ps[wave][pbase + i]);
            pa[8 + i] = us2bf(Ps[wave][pbase + 16 + i]);
        }
        __builtin_amdgcn_wave_barrier();

        // ---- batch-load all 8 V operands, then 8 WMMAs
        bf16x16 bv[8];
#pragma unroll
        for (int t = 0; t < 8; ++t) {
            const unsigned short* vp =
                Vp + (size_t)(t * 16 + row) * 2048 + kb + half * 8;
#pragma unroll
            for (int i = 0; i < 8; ++i) {
                bv[t][i]     = us2bf(vp[i]);
                bv[t][8 + i] = us2bf(vp[16 + i]);
            }
        }
#pragma unroll
        for (int t = 0; t < 8; ++t) ctx[t] = wmma_bf16(pa, bv[t], ctx[t]);
    }

    // ---- normalize + store ctx as [B, S, H*HD] bf16 (feeds proj GEMM)
    const int bi = bh >> 4, h = bh & 15;
#pragma unroll
    for (int r = 0; r < 8; ++r) {
        const float inv = 1.0f / lrow[r];
        const int m = q0 + r + half * 8;
#pragma unroll
        for (int t = 0; t < 8; ++t) {
            const size_t idx =
                ((size_t)bi * 2048 + m) * 2048 + h * 128 + t * 16 + row;
            Ctx[idx] = f2us_bf16(ctx[t][r] * inv);
        }
    }
}

// ---------------------------------------------------------------------------
// launch
// ---------------------------------------------------------------------------
extern "C" void kernel_launch(void* const* d_in, const int* in_sizes, int n_in,
                              void* d_out, int out_size, void* d_ws, size_t ws_size,
                              hipStream_t stream) {
    (void)in_sizes; (void)n_in; (void)out_size; (void)ws_size;

    const float* hs       = (const float*)d_in[0];  // [2,2048,2048]
    const float* c_attn_w = (const float*)d_in[1];  // [2048,6144]
    const float* c_attn_b = (const float*)d_in[2];  // [6144]
    const float* c_proj_w = (const float*)d_in[3];  // [2048,2048]
    const float* c_proj_b = (const float*)d_in[4];  // [2048]
    float* out = (float*)d_out;                     // [2,2048,2048]

    const size_t BSD = (size_t)2 * 2048 * 2048;     // 8388608
    unsigned short* ws    = (unsigned short*)d_ws;
    unsigned short* Xbf   = ws;
    unsigned short* Wqkv  = Xbf  + BSD;
    unsigned short* Wproj = Wqkv + (size_t)2048 * 6144;
    unsigned short* Qb    = Wproj + (size_t)2048 * 2048;
    unsigned short* Kb    = Qb + BSD;
    unsigned short* Vb    = Kb + BSD;   // stored transposed [B,H,HD,S]
    unsigned short* Ctx   = Vb + BSD;

    cvt_f32_bf16<<<4096, 256, 0, stream>>>(hs, Xbf, (int)BSD);
    cvt_f32_bf16<<<4096, 256, 0, stream>>>(c_attn_w, Wqkv, 2048 * 6144);
    cvt_f32_bf16<<<4096, 256, 0, stream>>>(c_proj_w, Wproj, 2048 * 2048);

    dim3 gqkv(6144 / 256, 4096 / 32);   // (24, 128)
    gemm_bf16<0><<<gqkv, 256, 0, stream>>>(Xbf, Wqkv, c_attn_b,
                                           4096, 2048, 6144,
                                           Qb, Kb, Vb, nullptr);

    attn_flash<<<512, 256, 0, stream>>>(Qb, Kb, Vb, Ctx);

    dim3 gproj(2048 / 256, 4096 / 32);  // (8, 128)
    gemm_bf16<1><<<gproj, 256, 0, stream>>>(Ctx, Wproj, c_proj_b,
                                            4096, 2048, 2048,
                                            nullptr, nullptr, nullptr, out);
}
